// HunyuanImage3Attention_78469052498385
// MI455X (gfx1250) — compile-verified, matrix operand order checked by reading, not run on hardware
//
#include <hip/hip_runtime.h>

typedef unsigned short u16;
typedef __attribute__((ext_vector_type(4)))  int    v4i;     // 16 bytes
typedef __attribute__((ext_vector_type(8)))  u16    u16x8;   // 16 bytes
typedef __attribute__((ext_vector_type(16))) __bf16 v16bf;   // 32 bytes (A/B frag)
typedef __attribute__((ext_vector_type(8)))  float  v8f;     // 32 bytes (C/D frag)

#define S_LEN   2048
#define DMODEL  4096
#define NQKV    6144
#define HKV     8
#define GRP     4
#define HD      128
#define MROWS   4096        // B*S
#define CK      64          // attention KV chunk

// gfx1250 async global->LDS DMA (ASYNCcnt-tracked), guarded for portability.
#if defined(__HIP_DEVICE_COMPILE__) && __has_builtin(__builtin_amdgcn_global_load_async_to_lds_b128)
#define HAVE_ASYNC 1
#else
#define HAVE_ASYNC 0
#endif

#if HAVE_ASYNC
typedef __attribute__((address_space(1))) v4i* gv4i_t;   // global, typed v4i
typedef __attribute__((address_space(3))) v4i* lv4i_t;   // LDS, typed v4i
__device__ inline void async_b128(const void* g, void* l) {
    __builtin_amdgcn_global_load_async_to_lds_b128((gv4i_t)g, (lv4i_t)l, 0, 0);
}
__device__ inline void wait_async0() {
    asm volatile("s_wait_asynccnt 0x0" ::: "memory");
}
#endif

__device__ inline u16 f2bf(float f) {
    unsigned int u = __float_as_uint(f);
    u += 0x7FFFu + ((u >> 16) & 1u);      // round-to-nearest-even
    return (u16)(u >> 16);
}

__device__ inline v8f vzero8() { v8f z; for (int i = 0; i < 8; ++i) z[i] = 0.0f; return z; }

union Frag { v16bf v; u16x8 h[2]; };

// Load one 16x32 bf16 A/B fragment from a K-contiguous [rows][ld] region.
// ISA layout: lane L -> row L%16; V0..3 hold K = 8*(L/16)+0..7, V4..7 hold K = 8*(L/16)+16..23.
__device__ inline void load_frag(Frag& f, const u16* base, int ld, int lane) {
    const int row = lane & 15;
    const int kb  = (lane >> 4) << 3;
    const u16* p = base + row * ld + kb;
    f.h[0] = *reinterpret_cast<const u16x8*>(p);
    f.h[1] = *reinterpret_cast<const u16x8*>(p + 16);
}

// ---------------------------------------------------------------- fp32 -> bf16
__global__ void __launch_bounds__(256) cvt_f32_bf16_k(const float* __restrict__ src,
                                                      u16* __restrict__ dst, int n) {
    int i = (blockIdx.x * 256 + threadIdx.x) * 4;
    if (i < n) {
        float4 f = *reinterpret_cast<const float4*>(src + i);
        dst[i + 0] = f2bf(f.x); dst[i + 1] = f2bf(f.y);
        dst[i + 2] = f2bf(f.z); dst[i + 3] = f2bf(f.w);
    }
}

// ------------------------------------------------- bf16 NT GEMM: C = A * B^T
// A[M][K], B[N][K] (both K-contiguous), C[M][N] fp32.
// Block tile 128(M) x 256(N), 8 waves as 2x4, 64x64 per wave (16 WMMAs/k-step).
// Double-buffered LDS, async DMA staging when available.
__global__ void __launch_bounds__(256) gemm_bf16_nt_k(const u16* __restrict__ A,
                                                      const u16* __restrict__ Bm,
                                                      float* __restrict__ C,
                                                      int M, int N, int K) {
    __shared__ u16 As[2][128 * 32];     // 2 x 8 KB
    __shared__ u16 Bs[2][256 * 32];     // 2 x 16 KB
    const int t = threadIdx.x, lane = t & 31, wave = t >> 5;
    const int m0 = blockIdx.y * 128, n0 = blockIdx.x * 256;
    const int wm = (wave >> 2) * 64;    // 0 / 64
    const int wn = (wave & 3) * 64;     // 0..192

    v8f acc[4][4];
    for (int i = 0; i < 4; ++i) for (int j = 0; j < 4; ++j) acc[i][j] = vzero8();

    // staging coordinates: A -> 2 x b128 / thread, B -> 4 x b128 / thread
    const int ar = t & 127, ah = (t >> 7) * 16;          // A row, k-half (elems)
    const u16* Ag0 = &A[(size_t)(m0 + ar) * K + ah];
    const u16* Bg0 = &Bm[(size_t)(n0 + t) * K];

#if HAVE_ASYNC
    // prologue: DMA tile 0 into buffer 0
    {
        const u16* ag = Ag0;            u16* al = &As[0][ar * 32 + ah];
        const u16* bg = Bg0;            u16* bl = &Bs[0][t * 32];
        async_b128(ag, al);             async_b128(ag + 8,  al + 8);
        async_b128(bg, bl);             async_b128(bg + 8,  bl + 8);
        async_b128(bg + 16, bl + 16);   async_b128(bg + 24, bl + 24);
    }
    int cur = 0;
    for (int kk = 0; kk < K; kk += 32) {
        wait_async0();                  // own tile-k DMA complete
        __syncthreads();                // tile k visible; alt buffer readers done
        if (kk + 32 < K) {              // DMA tile k+1 under the WMMAs below
            const u16* ag = Ag0 + kk + 32;  u16* al = &As[cur ^ 1][ar * 32 + ah];
            const u16* bg = Bg0 + kk + 32;  u16* bl = &Bs[cur ^ 1][t * 32];
            async_b128(ag, al);             async_b128(ag + 8,  al + 8);
            async_b128(bg, bl);             async_b128(bg + 8,  bl + 8);
            async_b128(bg + 16, bl + 16);   async_b128(bg + 24, bl + 24);
        }
        Frag a[4];
        for (int i = 0; i < 4; ++i) load_frag(a[i], &As[cur][(wm + i * 16) * 32], 32, lane);
        for (int j = 0; j < 4; ++j) {
            Frag b;
            load_frag(b, &Bs[cur][(wn + j * 16) * 32], 32, lane);
            for (int i = 0; i < 4; ++i)
                acc[i][j] = __builtin_amdgcn_wmma_f32_16x16x32_bf16(
                    false, a[i].v, false, b.v, (short)0, acc[i][j], false, false);
        }
        cur ^= 1;
    }
#else
    // synchronous fallback: VGPR staging, single buffer
    for (int kk = 0; kk < K; kk += 32) {
        u16x8 av0 = *reinterpret_cast<const u16x8*>(Ag0 + kk);
        u16x8 av1 = *reinterpret_cast<const u16x8*>(Ag0 + kk + 8);
        u16x8 bv[4];
        for (int u = 0; u < 4; ++u)
            bv[u] = *reinterpret_cast<const u16x8*>(Bg0 + kk + u * 8);
        *reinterpret_cast<u16x8*>(&As[0][ar * 32 + ah])     = av0;
        *reinterpret_cast<u16x8*>(&As[0][ar * 32 + ah + 8]) = av1;
        for (int u = 0; u < 4; ++u)
            *reinterpret_cast<u16x8*>(&Bs[0][t * 32 + u * 8]) = bv[u];
        __syncthreads();
        Frag a[4];
        for (int i = 0; i < 4; ++i) load_frag(a[i], &As[0][(wm + i * 16) * 32], 32, lane);
        for (int j = 0; j < 4; ++j) {
            Frag b;
            load_frag(b, &Bs[0][(wn + j * 16) * 32], 32, lane);
            for (int i = 0; i < 4; ++i)
                acc[i][j] = __builtin_amdgcn_wmma_f32_16x16x32_bf16(
                    false, a[i].v, false, b.v, (short)0, acc[i][j], false, false);
        }
        __syncthreads();
    }
#endif

    const int cr = (lane >> 4) << 3, cn = lane & 15;
    for (int i = 0; i < 4; ++i)
        for (int j = 0; j < 4; ++j)
            for (int r = 0; r < 8; ++r) {
                int row = m0 + wm + i * 16 + r + cr;
                int col = n0 + wn + j * 16 + cn;
                C[(size_t)row * N + col] = acc[i][j][r];
            }
}

// -------------------------------------- RoPE + RMS-norm + scatter to q/k/v bf16
// qkv fp32 laid out [b][s][kv][slot=0..5][128]; one wave per 128-vector.
__global__ void __launch_bounds__(256) rope_rms_scatter_k(
    const float* __restrict__ qkv, const float* __restrict__ cosb,
    const float* __restrict__ sinb, const float* __restrict__ qw,
    const float* __restrict__ kw, u16* __restrict__ qo,
    u16* __restrict__ ko, u16* __restrict__ vo) {
    const int lane = threadIdx.x & 31, wave = threadIdx.x >> 5;
    const int vid  = blockIdx.x * 8 + wave;
    const int slot = vid % 6;
    const int kv   = (vid / 6) & (HKV - 1);
    const int s    = (vid / 48) & (S_LEN - 1);
    const int b    = vid / (48 * S_LEN);

    const float* x = qkv + (size_t)vid * HD;
    const int d0 = lane, d1 = lane + 32, d2 = lane + 64, d3 = lane + 96;
    float x0 = x[d0], x1 = x[d1], x2 = x[d2], x3 = x[d3];

    if (slot == 5) {                                         // V: just cast
        size_t base = (((size_t)(b * HKV + kv)) * S_LEN + s) * HD;
        vo[base + d0] = f2bf(x0); vo[base + d1] = f2bf(x1);
        vo[base + d2] = f2bf(x2); vo[base + d3] = f2bf(x3);
        return;
    }
    const float* c  = cosb + (size_t)s * HD;
    const float* sn = sinb + (size_t)s * HD;
    float y0 = x0 * c[d0] - x2 * sn[d0];                     // rotate_half pairs
    float y1 = x1 * c[d1] - x3 * sn[d1];
    float y2 = x2 * c[d2] + x0 * sn[d2];
    float y3 = x3 * c[d3] + x1 * sn[d3];

    float ss = y0 * y0 + y1 * y1 + y2 * y2 + y3 * y3;
    for (int m = 16; m >= 1; m >>= 1) ss += __shfl_xor(ss, m, 32);
    float rn = rsqrtf(ss * (1.0f / 128.0f) + 1e-5f);

    const float* w = (slot == 4) ? kw : qw;
    y0 *= w[d0] * rn; y1 *= w[d1] * rn; y2 *= w[d2] * rn; y3 *= w[d3] * rn;

    if (slot == 4) {
        size_t base = (((size_t)(b * HKV + kv)) * S_LEN + s) * HD;
        ko[base + d0] = f2bf(y0); ko[base + d1] = f2bf(y1);
        ko[base + d2] = f2bf(y2); ko[base + d3] = f2bf(y3);
    } else {
        size_t base = (((size_t)(b * HKV + kv)) * (GRP * S_LEN) + slot * S_LEN + s) * HD;
        qo[base + d0] = f2bf(y0); qo[base + d1] = f2bf(y1);
        qo[base + d2] = f2bf(y2); qo[base + d3] = f2bf(y3);
    }
}

// ------------------------------------------------------------ flash attention
// grid = (64 q-tiles, B*HKV). Each wave: 16 q rows, streams 32 KV chunks of 64.
__global__ void __launch_bounds__(256) flash_attn_k(
    const u16* __restrict__ q, const u16* __restrict__ k,
    const u16* __restrict__ v, u16* __restrict__ o) {
    __shared__ u16 Ks[CK * HD];          // [key][dim]      16 KB
    __shared__ u16 Vt[HD * CK];          // [dim][key]      16 KB
    __shared__ u16 Ps[8 * 16 * CK];      // per-wave P tile 16 KB

    const int t = threadIdx.x, lane = t & 31, wave = t >> 5;
    const int bk = blockIdx.y;                 // b*HKV + kv
    const int kv = bk & (HKV - 1), b = bk >> 3;
    const int q_base = blockIdx.x * 128 + wave * 16;

    const u16* Qp = q + ((size_t)bk * (GRP * S_LEN) + q_base) * HD;
    const u16* Kp = k + (size_t)bk * S_LEN * HD;
    const u16* Vp = v + (size_t)bk * S_LEN * HD;

    Frag aQ[4];                                // Q fragments live in registers
    for (int c = 0; c < 4; ++c) load_frag(aQ[c], Qp + c * 32, HD, lane);

    float rowmax[8], rowsum[8];
    v8f accO[8];
    for (int r = 0; r < 8; ++r) { rowmax[r] = -1e30f; rowsum[r] = 0.0f; }
    for (int tt = 0; tt < 8; ++tt) accO[tt] = vzero8();
    const float scale = 0.08838834764831845f;  // 1/sqrt(128)

    const int lrow = t >> 2;                   // 0..63 (key row)
    const int lco  = (t & 3) * 32;             // dim offset

    for (int kc = 0; kc < S_LEN; kc += CK) {
        // stage K chunk [64][128]
        const u16* ksrc = Kp + (size_t)(kc + lrow) * HD + lco;
        for (int u = 0; u < 4; ++u)
            *reinterpret_cast<u16x8*>(&Ks[lrow * HD + lco + u * 8]) =
                *reinterpret_cast<const u16x8*>(ksrc + u * 8);
        // stage V chunk transposed -> [dim][key]
        const u16* vsrc = Vp + (size_t)(kc + lrow) * HD + lco;
        u16x8 vv[4];
        for (int u = 0; u < 4; ++u) vv[u] = *reinterpret_cast<const u16x8*>(vsrc + u * 8);
        for (int u = 0; u < 4; ++u)
            for (int e = 0; e < 8; ++e)
                Vt[(lco + u * 8 + e) * CK + lrow] = vv[u][e];
        __syncthreads();

        // S = Q K^T  (4 key subtiles x 4 k-steps)
        v8f sc[4];
        for (int j = 0; j < 4; ++j) sc[j] = vzero8();
        for (int j = 0; j < 4; ++j)
            for (int c = 0; c < 4; ++c) {
                Frag bK; load_frag(bK, &Ks[(j * 16) * HD + c * 32], HD, lane);
                sc[j] = __builtin_amdgcn_wmma_f32_16x16x32_bf16(
                    false, aQ[c].v, false, bK.v, (short)0, sc[j], false, false);
            }

        // online softmax on C-layout rows (row = r + 8*(lane/16))
        float corr[8];
        for (int r = 0; r < 8; ++r) {
            float mx = -1e30f;
            for (int j = 0; j < 4; ++j) { sc[j][r] *= scale; mx = fmaxf(mx, sc[j][r]); }
            for (int m = 8; m >= 1; m >>= 1) mx = fmaxf(mx, __shfl_xor(mx, m, 32));
            float nm = fmaxf(rowmax[r], mx);
            corr[r] = __expf(rowmax[r] - nm);
            rowmax[r] = nm;
            float ls = 0.0f;
            for (int j = 0; j < 4; ++j) { float p = __expf(sc[j][r] - nm); sc[j][r] = p; ls += p; }
            for (int m = 8; m >= 1; m >>= 1) ls += __shfl_xor(ls, m, 32);
            rowsum[r] = rowsum[r] * corr[r] + ls;
        }

        // P -> per-wave LDS (C-layout -> row-major [16][64] bf16)
        u16* Pw = &Ps[wave * 16 * CK];
        const int prow = (lane >> 4) << 3;
        for (int j = 0; j < 4; ++j)
            for (int r = 0; r < 8; ++r)
                Pw[(prow + r) * CK + j * 16 + (lane & 15)] = f2bf(sc[j][r]);

        // O = diag(corr) O + P V
        Frag aP[2];
        for (int c = 0; c < 2; ++c) load_frag(aP[c], Pw + c * 32, CK, lane);
        for (int tt = 0; tt < 8; ++tt) {
            for (int r = 0; r < 8; ++r) accO[tt][r] *= corr[r];
            for (int c = 0; c < 2; ++c) {
                Frag bV; load_frag(bV, &Vt[(tt * 16) * CK + c * 32], CK, lane);
                accO[tt] = __builtin_amdgcn_wmma_f32_16x16x32_bf16(
                    false, aP[c].v, false, bV.v, (short)0, accO[tt], false, false);
            }
        }
        __syncthreads();
    }

    // normalize + scatter to [b][s][head*128 + d] bf16
    const int cr = (lane >> 4) << 3, cn = lane & 15;
    for (int r = 0; r < 8; ++r) {
        int m  = r + cr;
        int qi = q_base + m;
        int g  = qi >> 11, s = qi & (S_LEN - 1);
        int head = kv * GRP + g;
        float inv = 1.0f / rowsum[r];
        size_t base = ((size_t)(b * S_LEN + s)) * DMODEL + head * HD + cn;
        for (int tt = 0; tt < 8; ++tt)
            o[base + tt * 16] = f2bf(accO[tt][r] * inv);
    }
}

// ------------------------------------------------------------------- launcher
extern "C" void kernel_launch(void* const* d_in, const int* in_sizes, int n_in,
                              void* d_out, int out_size, void* d_ws, size_t ws_size,
                              hipStream_t stream) {
    (void)in_sizes; (void)n_in; (void)out_size; (void)ws_size;
    const float* hs   = (const float*)d_in[0];
    const float* cosb = (const float*)d_in[1];
    const float* sinb = (const float*)d_in[2];
    const float* wqkv = (const float*)d_in[3];
    const float* wo   = (const float*)d_in[4];
    const float* qw   = (const float*)d_in[5];
    const float* kw   = (const float*)d_in[6];
    float* out = (float*)d_out;

    char* ws = (char*)d_ws;
    size_t off = 0;
    auto bump = [&](size_t bytes) -> void* {
        void* p = ws + off; off += (bytes + 255) & ~(size_t)255; return p;
    };
    u16*   hs_bf   = (u16*)  bump((size_t)MROWS * DMODEL * 2);   // 32 MB
    u16*   wqkv_bf = (u16*)  bump((size_t)NQKV * DMODEL * 2);    // 48 MB
    float* qkvf    = (float*)bump((size_t)MROWS * NQKV * 4);     // 100 MB
    u16*   q_bf    = (u16*)  bump((size_t)2 * HKV * GRP * S_LEN * HD * 2); // 32 MB
    u16*   k_bf    = (u16*)  bump((size_t)2 * HKV * S_LEN * HD * 2);       // 8 MB
    u16*   v_bf    = (u16*)  bump((size_t)2 * HKV * S_LEN * HD * 2);       // 8 MB
    u16*   ao_bf   = (u16*)qkvf;     // alias: qkv fp32 dead after rope_rms
    u16*   wo_bf   = hs_bf;          // alias: hidden bf16 dead after GEMM1

    // 1) casts for QKV GEMM
    {
        int n1 = MROWS * DMODEL;
        cvt_f32_bf16_k<<<n1 / 1024, 256, 0, stream>>>(hs, hs_bf, n1);
        int n2 = NQKV * DMODEL;
        cvt_f32_bf16_k<<<n2 / 1024, 256, 0, stream>>>(wqkv, wqkv_bf, n2);
    }
    // 2) QKV projection: [4096,4096] x [6144,4096]^T -> fp32
    {
        dim3 g(NQKV / 256, MROWS / 128);
        gemm_bf16_nt_k<<<g, 256, 0, stream>>>(hs_bf, wqkv_bf, qkvf, MROWS, NQKV, DMODEL);
    }
    // 3) w_o cast (into the now-dead hidden-bf16 region)
    {
        int n3 = DMODEL * DMODEL;
        cvt_f32_bf16_k<<<n3 / 1024, 256, 0, stream>>>(wo, wo_bf, n3);
    }
    // 4) RoPE + RMS + scatter (196608 vectors, 8 waves/block)
    rope_rms_scatter_k<<<(2 * S_LEN * HKV * 6) / 8, 256, 0, stream>>>(
        qkvf, cosb, sinb, qw, kw, q_bf, k_bf, v_bf);
    // 5) attention
    {
        dim3 g((GRP * S_LEN) / 128, 2 * HKV);
        flash_attn_k<<<g, 256, 0, stream>>>(q_bf, k_bf, v_bf, ao_bf);
    }
    // 6) output projection -> fp32 d_out
    {
        dim3 g(DMODEL / 256, MROWS / 128);
        gemm_bf16_nt_k<<<g, 256, 0, stream>>>(ao_bf, wo_bf, out, MROWS, DMODEL, DMODEL);
    }
}
